// PINNEdgePredictor_32882269618485
// MI455X (gfx1250) — matrix operand order repaired and checked by reading, not compiled, not run on hardware
//
#include <hip/hip_runtime.h>
#include <hip/hip_bf16.h>

// ---------------------------------------------------------------------------
// PINN edge predictor, fully fused per-graph kernel for gfx1250 (MI455X).
// One workgroup (128 threads = 4 wave32) per graph. Whole pipeline in LDS.
// ---------------------------------------------------------------------------

typedef __attribute__((ext_vector_type(16))) _Float16 v16h;
typedef __attribute__((ext_vector_type(8)))  float    v8f;

#define NPG 57      // nodes per graph
#define NBR 80      // branches per graph
#define HID 128
#define NL  5
#define EPG 160     // bidirectional edges per graph

// ---- LDS layout (bytes) ---------------------------------------------------
// Phase A (GCN):   hs[64*128]f32 | ha[64*128]f16 | Wt[128*128]f16 |
//                  hw[64*128]f32 | acc[64*128]f32 | misc
// Phase B (edges): hs (until eh built) | eh[80*256]f16 (over ha+Wt) |
//                  W1t[128*256]f16 (over hw+acc) | misc
#define OFF_HS    0
#define OFF_HA    32768
#define OFF_WT    49152
#define OFF_HW    81920
#define OFF_ACC   114688
#define OFF_EH    32768     // 40960 B, overlays ha+Wt (dead in phase B)
#define OFF_W1T   81920     // 65536 B, overlays hw+acc (dead in phase B)
#define OFF_MISC  147456
// misc sub-offsets (bytes from OFF_MISC)
#define MO_DINV   0         // f32[64]
#define MO_THETA  256       // f32[64]
#define MO_NORM   512       // f32[160]
#define MO_ESRC   1152      // i32[160]
#define MO_EDST   1792      // i32[160]
#define MO_PW1    2432      // f32[2048]  phys_W1
#define MO_PB1    10624     // f32[16]
#define MO_PW2    10688     // f32[16]
#define MO_W1L    10752     // f32[128]   mlp_W1 row 256
#define MO_W2     11264     // f32[128]   mlp_W2
#define MO_B1     11776     // f32[128]   mlp_b1
#define MO_DTH    12288     // f32[80]
#define MO_BU     12608     // i32[80]
#define MO_BV     12928     // i32[80]
#define MO_XL     13248     // f32[228]
#define MISC_BYTES 14160
#define SMEM_BYTES (OFF_MISC + MISC_BYTES)   // 161616 B < 320 KB/WGP

union FragU { v16h v; float4 f[2]; };

// A fragment (16x32 f16), row-major source with `stride` halves per row.
// ISA layout: lane L holds row r0+(L&15); halves 0..7 = K (k0+8*(L>>4))..+7,
// halves 8..15 = same +16.  -> two contiguous 16B LDS loads.
__device__ __forceinline__ v16h lds_load_a16(const _Float16* base, int stride,
                                             int row0, int k0, int lane) {
  const _Float16* p = base + (row0 + (lane & 15)) * stride + k0 + 8 * (lane >> 4);
  FragU u;
  u.f[0] = *reinterpret_cast<const float4*>(p);
  u.f[1] = *reinterpret_cast<const float4*>(p + 16);
  return u.v;
}

// B fragment (32x16 f16) from K-contiguous (transposed) weights BT[n][k].
// ISA layout: lane L holds col n0+(L&15); halves 0..7 = K (k0+8*(L>>4))..+7,
// halves 8..15 = same +16.
__device__ __forceinline__ v16h lds_load_b16(const _Float16* baseT, int stride,
                                             int n0, int k0, int lane) {
  const _Float16* p = baseT + (n0 + (lane & 15)) * stride + k0 + 8 * (lane >> 4);
  FragU u;
  u.f[0] = *reinterpret_cast<const float4*>(p);
  u.f[1] = *reinterpret_cast<const float4*>(p + 16);
  return u.v;
}

__global__ __launch_bounds__(128)
void pinn_edge_fused_kernel(const float* __restrict__ x,
                            const int*   __restrict__ edge_index,
                            const int*   __restrict__ branch_u,
                            const int*   __restrict__ branch_v,
                            const float* __restrict__ Wp,
                            const float* __restrict__ bp,
                            const float* __restrict__ conv_W,
                            const float* __restrict__ conv_b,
                            const float* __restrict__ bn_gamma,
                            const float* __restrict__ bn_beta,
                            const float* __restrict__ bn_mean,
                            const float* __restrict__ bn_var,
                            const float* __restrict__ phys_W1,
                            const float* __restrict__ phys_b1,
                            const float* __restrict__ phys_W2,
                            const float* __restrict__ phys_b2,
                            const float* __restrict__ mlp_W1,
                            const float* __restrict__ mlp_b1,
                            const float* __restrict__ mlp_W2,
                            const float* __restrict__ mlp_b2,
                            float* __restrict__ out,
                            int E_total) {
  extern __shared__ __align__(16) char smem[];
  float*    hs   = reinterpret_cast<float*>(smem + OFF_HS);     // h (f32) [64][128]
  _Float16* ha   = reinterpret_cast<_Float16*>(smem + OFF_HA);  // h (f16) [64][128]
  _Float16* Wt   = reinterpret_cast<_Float16*>(smem + OFF_WT);  // W^T f16 [128][128]
  float*    hw   = reinterpret_cast<float*>(smem + OFF_HW);     // h@W [64][128]
  float*    acc  = reinterpret_cast<float*>(smem + OFF_ACC);    // aggregated [64][128]
  _Float16* eh   = reinterpret_cast<_Float16*>(smem + OFF_EH);  // [80][256] f16
  _Float16* W1t  = reinterpret_cast<_Float16*>(smem + OFF_W1T); // [128][256] f16
  float* dinv    = reinterpret_cast<float*>(smem + OFF_MISC + MO_DINV);
  float* theta   = reinterpret_cast<float*>(smem + OFF_MISC + MO_THETA);
  float* norms   = reinterpret_cast<float*>(smem + OFF_MISC + MO_NORM);
  int*   esrc    = reinterpret_cast<int*>(smem + OFF_MISC + MO_ESRC);
  int*   edst    = reinterpret_cast<int*>(smem + OFF_MISC + MO_EDST);
  float* pW1s    = reinterpret_cast<float*>(smem + OFF_MISC + MO_PW1);
  float* pb1s    = reinterpret_cast<float*>(smem + OFF_MISC + MO_PB1);
  float* pW2s    = reinterpret_cast<float*>(smem + OFF_MISC + MO_PW2);
  float* w1last  = reinterpret_cast<float*>(smem + OFF_MISC + MO_W1L);
  float* w2f     = reinterpret_cast<float*>(smem + OFF_MISC + MO_W2);
  float* b1f     = reinterpret_cast<float*>(smem + OFF_MISC + MO_B1);
  float* dth     = reinterpret_cast<float*>(smem + OFF_MISC + MO_DTH);
  int*   bu      = reinterpret_cast<int*>(smem + OFF_MISC + MO_BU);
  int*   bv      = reinterpret_cast<int*>(smem + OFF_MISC + MO_BV);
  float* xl      = reinterpret_cast<float*>(smem + OFF_MISC + MO_XL);

  const int tid  = threadIdx.x;
  const int lane = tid & 31;
  const int wave = tid >> 5;
  const int g    = blockIdx.x;
  const int nodebase = g * NPG;

  // ---- stage inputs for this graph -------------------------------------
  for (int i = tid; i < NPG * 4; i += 128) xl[i] = x[nodebase * 4 + i];
  for (int j = tid; j < EPG; j += 128) {
    esrc[j] = edge_index[g * EPG + j] - nodebase;
    edst[j] = edge_index[E_total + g * EPG + j] - nodebase;
  }
  for (int i = tid; i < HID * 16; i += 128) pW1s[i] = phys_W1[i];
  if (tid < 16) { pb1s[tid] = phys_b1[tid]; pW2s[tid] = phys_W2[tid]; }
  for (int i = tid; i < NBR; i += 128) { bu[i] = branch_u[i]; bv[i] = branch_v[i]; }
  __syncthreads();

  // ---- degrees (self-loop included) & edge norms -----------------------
  if (tid < NPG) {
    int c = 1;
    for (int j = 0; j < EPG; ++j) c += (edst[j] == tid) ? 1 : 0;
    dinv[tid] = rsqrtf((float)c);
  } else if (tid < 64) {
    dinv[tid] = 0.f;
  }
  __syncthreads();
  for (int j = tid; j < EPG; j += 128) norms[j] = dinv[esrc[j]] * dinv[edst[j]];

  // ---- input projection: h0 = x @ Wp + bp (thread owns channel tid) ----
  {
    const int c = tid;
    const float w0 = Wp[c], w1 = Wp[HID + c], w2 = Wp[2 * HID + c], w3 = Wp[3 * HID + c];
    const float bb = bp[c];
    for (int n = 0; n < NPG; ++n) {
      float v = bb + xl[n * 4] * w0 + xl[n * 4 + 1] * w1 +
                xl[n * 4 + 2] * w2 + xl[n * 4 + 3] * w3;
      hs[n * HID + c] = v;
      ha[n * HID + c] = (_Float16)v;
    }
    for (int n = NPG; n < 64; ++n) { hs[n * HID + c] = 0.f; ha[n * HID + c] = (_Float16)0.f; }
  }
  __syncthreads();

  // ---- 5 GCN layers -----------------------------------------------------
  for (int l = 0; l < NL; ++l) {
    // stage W^T (K-contiguous) in f16
    const float* Wl = conv_W + l * HID * HID;
    for (int k = 0; k < HID; ++k)
      Wt[tid * HID + k] = (_Float16)Wl[k * HID + tid];
    if (l + 1 < NL)
      __builtin_prefetch(conv_W + (l + 1) * HID * HID + tid, 0, 1);
    __syncthreads();

    // GEMM: hw[64][128] = ha[64][128] @ W[128][128], WMMA 16x16x32
    {
      const int rt = wave;                 // row tile, rows rt*16..rt*16+15
      v16h afr[4];
#pragma unroll
      for (int kk = 0; kk < 4; ++kk)
        afr[kk] = lds_load_a16(ha, HID, rt * 16, kk * 32, lane);
      const int row0 = rt * 16 + 8 * (lane >> 4);
      const int coll = lane & 15;
#pragma unroll
      for (int ct = 0; ct < 8; ++ct) {
        v8f c = {};
#pragma unroll
        for (int kk = 0; kk < 4; ++kk) {
          v16h bf = lds_load_b16(Wt, HID, ct * 16, kk * 32, lane);
          c = __builtin_amdgcn_wmma_f32_16x16x32_f16(
              false, afr[kk], false, bf, (short)0, c, false, false);
        }
#pragma unroll
        for (int r = 0; r < 8; ++r)
          hw[(row0 + r) * HID + ct * 16 + coll] = c[r];
      }
    }
    __syncthreads();

    // Aggregation (deterministic: each thread owns channel c) + BN/ReLU/res
    {
      const int c = tid;
      for (int n = 0; n < NPG; ++n)
        acc[n * HID + c] = dinv[n] * dinv[n] * hw[n * HID + c];
      for (int j = 0; j < EPG; ++j)
        acc[edst[j] * HID + c] += norms[j] * hw[esrc[j] * HID + c];

      const float mean = bn_mean[l * HID + c];
      const float rs   = rsqrtf(bn_var[l * HID + c] + 1e-5f);
      const float gam  = bn_gamma[l * HID + c];
      const float bet  = bn_beta[l * HID + c];
      const float cb   = conv_b[l * HID + c];
      for (int n = 0; n < NPG; ++n) {
        float v = (acc[n * HID + c] + cb - mean) * rs * gam + bet;
        v = fmaxf(v, 0.f);
        float hn = v + hs[n * HID + c];
        hs[n * HID + c] = hn;
        ha[n * HID + c] = (_Float16)hn;
      }
    }
    __syncthreads();
  }

  // ---- physics head: theta[n] = relu(h@W1+b1)@W2 + b2 ------------------
  if (tid < NPG) {
    float s = 0.f;
    for (int j = 0; j < 16; ++j) {
      float a = pb1s[j];
      for (int c = 0; c < HID; ++c) a += hs[tid * HID + c] * pW1s[c * 16 + j];
      s += fmaxf(a, 0.f) * pW2s[j];
    }
    theta[tid] = s + phys_b2[0];
  }
  if (tid < HID) {
    w1last[tid] = mlp_W1[256 * HID + tid];
    w2f[tid]    = mlp_W2[tid];
    b1f[tid]    = mlp_b1[tid];
  }
  __syncthreads();

  // ---- build edge-MLP operands (overlays GCN scratch) ------------------
  for (int i = tid; i < NBR; i += 128) dth[i] = theta[bu[i]] - theta[bv[i]];
  for (int k = 0; k < 256; ++k)                     // W1^T, K-contiguous f16
    W1t[tid * 256 + k] = (_Float16)mlp_W1[k * HID + tid];
  for (int idx = tid; idx < NBR * 256; idx += 128) { // A = [h[u] | h[v]] f16
    int i = idx >> 8, k = idx & 255;
    int node = (k < HID) ? bu[i] : bv[i];
    eh[idx] = (_Float16)hs[node * HID + (k & 127)];
  }
  __syncthreads();

  // ---- edge MLP: relu(A@W1 + b1 + dth*w1last) @ W2 + b2 ----------------
  const float b2v = mlp_b2[0];
  const int coll = lane & 15;
  const int grp  = lane >> 4;
  for (int rt = wave; rt < 5; rt += 4) {            // 5 row tiles of 16 rows
    float partial[8];
#pragma unroll
    for (int r = 0; r < 8; ++r) partial[r] = 0.f;
#pragma unroll
    for (int ct = 0; ct < 8; ++ct) {
      v8f c = {};
#pragma unroll
      for (int kk = 0; kk < 8; ++kk) {
        v16h af = lds_load_a16(eh, 256, rt * 16, kk * 32, lane);
        v16h bf = lds_load_b16(W1t, 256, ct * 16, kk * 32, lane);
        c = __builtin_amdgcn_wmma_f32_16x16x32_f16(
            false, af, false, bf, (short)0, c, false, false);
      }
#pragma unroll
      for (int r = 0; r < 8; ++r) {
        int row = rt * 16 + r + 8 * grp;
        int col = ct * 16 + coll;
        float v = c[r] + b1f[col] + dth[row] * w1last[col];
        v = fmaxf(v, 0.f);
        partial[r] += v * w2f[col];
      }
    }
#pragma unroll
    for (int r = 0; r < 8; ++r) {                   // reduce 16 N-lanes
      float s = partial[r];
      s += __shfl_xor(s, 1, 16);
      s += __shfl_xor(s, 2, 16);
      s += __shfl_xor(s, 4, 16);
      s += __shfl_xor(s, 8, 16);
      if (coll == 0) {
        int row = rt * 16 + r + 8 * grp;
        out[g * NBR + row] = s + b2v;
      }
    }
  }
}

extern "C" void kernel_launch(void* const* d_in, const int* in_sizes, int n_in,
                              void* d_out, int out_size, void* d_ws, size_t ws_size,
                              hipStream_t stream) {
  (void)n_in; (void)d_ws; (void)ws_size; (void)out_size;
  const float* x        = (const float*)d_in[0];
  const int*   ei       = (const int*)d_in[1];
  const int*   bu       = (const int*)d_in[2];
  const int*   bvp      = (const int*)d_in[3];
  const float* Wp       = (const float*)d_in[4];
  const float* bp       = (const float*)d_in[5];
  const float* conv_W   = (const float*)d_in[6];
  const float* conv_b   = (const float*)d_in[7];
  const float* bn_gamma = (const float*)d_in[8];
  const float* bn_beta  = (const float*)d_in[9];
  const float* bn_mean  = (const float*)d_in[10];
  const float* bn_var   = (const float*)d_in[11];
  const float* phys_W1  = (const float*)d_in[12];
  const float* phys_b1  = (const float*)d_in[13];
  const float* phys_W2  = (const float*)d_in[14];
  const float* phys_b2  = (const float*)d_in[15];
  const float* mlp_W1   = (const float*)d_in[16];
  const float* mlp_b1   = (const float*)d_in[17];
  const float* mlp_W2   = (const float*)d_in[18];
  const float* mlp_b2   = (const float*)d_in[19];
  float* out = (float*)d_out;

  const int num_graphs = in_sizes[0] / (NPG * 4);
  const int E_total    = in_sizes[1] / 2;

  hipFuncSetAttribute((const void*)pinn_edge_fused_kernel,
                      hipFuncAttributeMaxDynamicSharedMemorySize, SMEM_BYTES);

  pinn_edge_fused_kernel<<<dim3(num_graphs), dim3(128), SMEM_BYTES, stream>>>(
      x, ei, bu, bvp, Wp, bp, conv_W, conv_b, bn_gamma, bn_beta, bn_mean,
      bn_var, phys_W1, phys_b1, phys_W2, phys_b2, mlp_W1, mlp_b1, mlp_W2,
      mlp_b2, out, E_total);
}
